// MattesMI_45973329936800
// MI455X (gfx1250) — matrix operand order repaired and checked by reading, not compiled, not run on hardware
//
#include <hip/hip_runtime.h>
#include <hip/hip_bf16.h>

#define EPSF 1e-8f
#define BINS 64
#define HWIN 6                 // Gaussian half-window: weights beyond 6.5 bins < 1e-9
#define WIN  (2 * HWIN + 1)    // 13 bins

typedef __attribute__((ext_vector_type(16))) _Float16 v16h;
typedef __attribute__((ext_vector_type(8)))  float    v8f;

// Order-preserving float <-> uint encoding so atomicMin/Max on unsigned give float min/max.
__device__ __forceinline__ unsigned fenc(float f) {
  unsigned u = __float_as_uint(f);
  return (u & 0x80000000u) ? ~u : (u | 0x80000000u);
}
__device__ __forceinline__ float fdec(unsigned e) {
  unsigned u = (e & 0x80000000u) ? (e & 0x7FFFFFFFu) : ~e;
  return __uint_as_float(u);
}

// ---------------------------------------------------------------------------
// Kernel 0: init workspace (minmax slots + joint accumulator)
// ---------------------------------------------------------------------------
__global__ void mi_init(unsigned* __restrict__ wsu, float* __restrict__ joint_g) {
  int i = blockIdx.x * blockDim.x + threadIdx.x;
  if (i < 4) wsu[i] = (i & 1) ? 0u : 0xFFFFFFFFu;  // even: min (init max), odd: max (init 0)
  for (int j = i; j < BINS * BINS; j += gridDim.x * blockDim.x) joint_g[j] = 0.0f;
}

// ---------------------------------------------------------------------------
// Kernel 1: global min/max of both volumes
// ---------------------------------------------------------------------------
__global__ __launch_bounds__(256) void mi_minmax(const float* __restrict__ f,
                                                 const float* __restrict__ m,
                                                 unsigned* __restrict__ wsu, int n) {
  __shared__ unsigned s0, s1, s2, s3;
  int tid = threadIdx.x;
  if (tid == 0) { s0 = 0xFFFFFFFFu; s1 = 0u; s2 = 0xFFFFFFFFu; s3 = 0u; }
  __syncthreads();
  float fmn = 3.402823466e38f, fmx = -3.402823466e38f;
  float mmn = 3.402823466e38f, mmx = -3.402823466e38f;
  for (int i = blockIdx.x * blockDim.x + tid; i < n; i += gridDim.x * blockDim.x) {
    float a = f[i], b = m[i];
    fmn = fminf(fmn, a); fmx = fmaxf(fmx, a);
    mmn = fminf(mmn, b); mmx = fmaxf(mmx, b);
  }
  atomicMin(&s0, fenc(fmn)); atomicMax(&s1, fenc(fmx));
  atomicMin(&s2, fenc(mmn)); atomicMax(&s3, fenc(mmx));
  __syncthreads();
  if (tid == 0) {
    atomicMin(&wsu[0], s0); atomicMax(&wsu[1], s1);
    atomicMin(&wsu[2], s2); atomicMax(&wsu[3], s3);
  }
}

// ---------------------------------------------------------------------------
// Kernel 2: Parzen weights + joint histogram via V_WMMA_F32_16X16X32_F16.
// Each wave consumes 32 voxels per step (one K=32 slab) and owns the full
// 64x64 output as 4x4 tiles of 16x16 (16 v8f accumulators = 128 VGPRs).
// Only a 13-bin window around round(x_scaled) is materialized (compact
// Gaussian support); the rest of the f16 stage is zero-filled with b128 stores.
// ---------------------------------------------------------------------------
#define WAVES_PER_BLK 4

__global__ __launch_bounds__(32 * WAVES_PER_BLK) void mi_joint(
    const float* __restrict__ fixed, const float* __restrict__ moving,
    const unsigned* __restrict__ wsu, float* __restrict__ joint_g, int n) {
  __shared__ float ldsJoint[BINS * BINS];                                   // 16 KB
  __shared__ __align__(16) _Float16 stage[WAVES_PER_BLK][2][BINS * 32];     // 32 KB

  const int tid  = threadIdx.x;
  const int wave = tid >> 5;
  const int lane = tid & 31;
  const int half = lane >> 4;   // lane group (0: lanes 0-15, 1: lanes 16-31)
  const int mrow = lane & 15;

  for (int i = tid; i < BINS * BINS; i += blockDim.x) ldsJoint[i] = 0.0f;
  __syncthreads();

  const float fmin = fdec(wsu[0]), fmax = fdec(wsu[1]);
  const float mmin = fdec(wsu[2]), mmax = fdec(wsu[3]);
  const float fscale = (float)(BINS - 1) / (fmax - fmin + EPSF);
  const float mscale = (float)(BINS - 1) / (mmax - mmin + EPSF);

  v8f acc[16] = {};

  _Float16* fst = stage[wave][0];
  _Float16* mst = stage[wave][1];
  float4*   zst = (float4*)stage[wave];   // 8192 B = 512 float4s per wave

  const int waveGlobal = blockIdx.x * WAVES_PER_BLK + wave;
  const int nWaves     = gridDim.x * WAVES_PER_BLK;
  const int nchunk     = n >> 5;  // 32 voxels per chunk

  for (int c = waveGlobal; c < nchunk; c += nWaves) {
    // Zero both staging images: 16 x ds_store_b128 across the wave.
    const float4 z4 = make_float4(0.0f, 0.0f, 0.0f, 0.0f);
    #pragma unroll
    for (int i = 0; i < 16; ++i) zst[i * 32 + lane] = z4;
    asm volatile("" ::: "memory");

    const int vox = (c << 5) + lane;
    const float xf = (fixed[vox]  - fmin) * fscale;   // scaled to [0, 63]
    const float xm = (moving[vox] - mmin) * mscale;
    const int ibf = (int)(xf + 0.5f);
    const int ibm = (int)(xm + 0.5f);

    // 13-bin Gaussian windows; moving weights stored unnormalized,
    // fixed weights kept in registers until both normalizers are known.
    float wfv[WIN];
    float sf = 0.0f, sm = 0.0f;
    #pragma unroll
    for (int j = 0; j < WIN; ++j) {
      const int bF = ibf - HWIN + j;
      const int bM = ibm - HWIN + j;
      const float dF = xf - (float)bF;
      const float dM = xm - (float)bM;
      float wF = __expf(-0.5f * dF * dF);
      float wM = __expf(-0.5f * dM * dM);
      const bool vF = (unsigned)bF < (unsigned)BINS;
      const bool vM = (unsigned)bM < (unsigned)BINS;
      wF = vF ? wF : 0.0f;
      wfv[j] = wF; sf += wF;
      if (vM) mst[bM * 32 + lane] = (_Float16)wM;
      sm += vM ? wM : 0.0f;
    }
    // joint = sum_n (wf_n/(sf_n+eps)) x (wm_n/(sm_n+eps)): fold both into fixed.
    const float s = 1.0f / ((sf + EPSF) * (sm + EPSF));
    #pragma unroll
    for (int j = 0; j < WIN; ++j) {
      const int bF = ibf - HWIN + j;
      if ((unsigned)bF < (unsigned)BINS) fst[bF * 32 + lane] = (_Float16)(wfv[j] * s);
    }
    asm volatile("" ::: "memory");  // keep stage writes before cross-lane fragment reads

    // Build WMMA fragments straight from LDS (contiguous half pairs -> b128 loads).
    // A (16x32 f16): V0..3 hold K=2v,2v+1 (+8 for lanes 16-31); V4..7 hold K=16+2(v-4),.. (+8).
    // B (32x16 f16): same K layout as A: V0..3 hold K=2v,2v+1 (+8 for lanes 16-31); V4..7 hold K=16+2(v-4),.. (+8).
    union Frag { v16h v; unsigned u[8]; };
    Frag A[4], B[4];
    #pragma unroll
    for (int v = 0; v < 8; ++v) {
      const int ka = ((v & 4) ? 16 : 0) + 2 * (v & 3) + 8 * half;
      const int kb = ((v & 4) ? 16 : 0) + 2 * (v & 3) + 8 * half;
      #pragma unroll
      for (int t = 0; t < 4; ++t) {
        const int bin = 16 * t + mrow;
        A[t].u[v] = *(const unsigned*)&fst[bin * 32 + ka];
        B[t].u[v] = *(const unsigned*)&mst[bin * 32 + kb];
      }
    }
    asm volatile("" ::: "memory");  // keep fragment reads before next iter's stage writes

    // 16 WMMAs: 4x4 tiles of the 64x64 joint histogram, K = 32 voxels.
    #pragma unroll
    for (int mt = 0; mt < 4; ++mt)
      #pragma unroll
      for (int nt = 0; nt < 4; ++nt)
        acc[mt * 4 + nt] = __builtin_amdgcn_wmma_f32_16x16x32_f16(
            false, A[mt].v, false, B[nt].v, (short)0, acc[mt * 4 + nt], false, false);
  }

  // Wave-private accumulators -> block LDS joint (C/D layout: VGPR r => M=r+8*half, N=mrow).
  #pragma unroll
  for (int mt = 0; mt < 4; ++mt)
    #pragma unroll
    for (int nt = 0; nt < 4; ++nt)
      #pragma unroll
      for (int r = 0; r < 8; ++r) {
        const int row = 16 * mt + r + 8 * half;
        const int col = 16 * nt + mrow;
        atomicAdd(&ldsJoint[row * BINS + col], acc[mt * 4 + nt][r]);
      }
  __syncthreads();

  for (int i = tid; i < BINS * BINS; i += blockDim.x)
    atomicAdd(&joint_g[i], ldsJoint[i]);
}

// ---------------------------------------------------------------------------
// Kernel 3: normalize joint, marginals, MI reduction -> d_out[0] = -mi
// ---------------------------------------------------------------------------
__global__ __launch_bounds__(256) void mi_finalize(const float* __restrict__ joint_g,
                                                   float* __restrict__ out) {
  __shared__ float jn[BINS * BINS];
  __shared__ float pf[BINS], pm[BINS];
  __shared__ float red[256];
  const int tid = threadIdx.x;

  float part = 0.0f;
  for (int i = tid; i < BINS * BINS; i += 256) { float v = joint_g[i]; jn[i] = v; part += v; }
  red[tid] = part; __syncthreads();
  for (int off = 128; off > 0; off >>= 1) { if (tid < off) red[tid] += red[tid + off]; __syncthreads(); }
  const float inv = 1.0f / (red[0] + EPSF);

  for (int i = tid; i < BINS * BINS; i += 256) jn[i] *= inv;
  __syncthreads();

  if (tid < BINS) {
    float s = 0.0f;
    for (int j = 0; j < BINS; ++j) s += jn[tid * BINS + j];
    pf[tid] = s;
  } else if (tid < 2 * BINS) {
    const int j = tid - BINS;
    float s = 0.0f;
    for (int i = 0; i < BINS; ++i) s += jn[i * BINS + j];
    pm[j] = s;
  }
  __syncthreads();

  float mi = 0.0f;
  for (int c = tid; c < BINS * BINS; c += 256) {
    const float v = jn[c];
    if (v > EPSF) {
      const int i = c >> 6, j = c & 63;
      mi += v * __logf(v / (pf[i] * pm[j] + EPSF) + EPSF);
    }
  }
  red[tid] = mi; __syncthreads();
  for (int off = 128; off > 0; off >>= 1) { if (tid < off) red[tid] += red[tid + off]; __syncthreads(); }
  if (tid == 0) out[0] = -red[0];
}

// ---------------------------------------------------------------------------
extern "C" void kernel_launch(void* const* d_in, const int* in_sizes, int n_in,
                              void* d_out, int out_size, void* d_ws, size_t ws_size,
                              hipStream_t stream) {
  const float* fixedp  = (const float*)d_in[0];
  const float* movingp = (const float*)d_in[1];
  const int n = in_sizes[0];

  unsigned* wsu    = (unsigned*)d_ws;                 // [0..3]: enc(min/max) fixed, moving
  float*   joint_g = (float*)((char*)d_ws + 64);      // 64x64 f32 joint accumulator
  float*   outp    = (float*)d_out;

  mi_init   <<<16,  256, 0, stream>>>(wsu, joint_g);
  mi_minmax <<<512, 256, 0, stream>>>(fixedp, movingp, wsu, n);
  mi_joint  <<<256, 32 * WAVES_PER_BLK, 0, stream>>>(fixedp, movingp, wsu, joint_g, n);
  mi_finalize<<<1,  256, 0, stream>>>(joint_g, outp);
}
